// WHISPAblated_10660108828905
// MI455X (gfx1250) — matrix operand order repaired and checked by reading, not compiled, not run on hardware
//
#include <hip/hip_runtime.h>

typedef __attribute__((ext_vector_type(16))) _Float16 v16h;
typedef __attribute__((ext_vector_type(2)))  _Float16 h2;
typedef __attribute__((ext_vector_type(8)))  float    v8f;

namespace {
constexpr int   kD   = 8;
constexpr int   kNE  = 4;
constexpr int   kNS  = 3;
constexpr int   kNI  = 5;
constexpr float kEPS = 1e-5f;
constexpr int   kWavesPerBlock = 8;
constexpr int   kElemsPerWave  = 16;
constexpr int   kElemsPerBlock = kWavesPerBlock * kElemsPerWave; // 128
}

__global__ __launch_bounds__(256) void whisp_fused(
    const float* __restrict__ cl,   const float* __restrict__ cd,
    const float* __restrict__ relg, const float* __restrict__ mach,
    const float* __restrict__ alph, const float* __restrict__ u,
    const float* __restrict__ We,   const float* __restrict__ be,
    const float* __restrict__ Bmat, const float* __restrict__ Wproj,
    const float* __restrict__ bproj,const float* __restrict__ Wr1,
    const float* __restrict__ br1,  const float* __restrict__ Wr2,
    const float* __restrict__ br2,  const float* __restrict__ ln_g,
    const float* __restrict__ ln_b, const float* __restrict__ Wout,
    const float* __restrict__ bout, const float* __restrict__ fg,
    const float* __restrict__ fb,   const float* __restrict__ Wcst,
    const float* __restrict__ bcst, const float* __restrict__ Wcl,
    const float* __restrict__ bcl,  float* __restrict__ out, int B)
{
  // Cs[s][i][d] = sum_k Bm[s,i,k] * Wp[s,d,k] + bp[s,d]  (folded into WMMA accumulator)
  __shared__ float csbp[kNS][kNE][kD];
  {
    int t = threadIdx.x;
    if (t < kNS * kNE * kD) {
      int s = t / (kNE * kD), i = (t / kD) % kNE, d = t % kD;
      const float* bm = Bmat  + (s * kNE + i) * (kD * kD);
      const float* wp = Wproj + (s * kD  + d) * (kD * kD);
      float acc = bproj[s * kD + d];
      #pragma unroll
      for (int k = 0; k < kD * kD; ++k) acc = fmaf(bm[k], wp[k], acc);
      csbp[s][i][d] = acc;
    }
  }
  __syncthreads();

  const int lane = threadIdx.x & 31;
  const int wave = threadIdx.x >> 5;
  const int col  = lane & 15;   // element slot; both half-waves mirror the same element
  const int hi   = lane >> 4;   // 0 = lanes 0-15, 1 = lanes 16-31
  const int elem = blockIdx.x * kElemsPerBlock + wave * kElemsPerWave + col;
  if (elem >= B) return;        // never divergent for B % 128 == 0

  // ---------------- inputs (duplicated across half-waves; L0-cached) ----------------
  float uu[kD];
  #pragma unroll
  for (int d = 0; d < kD; ++d) uu[d] = u[elem * kD + d];
  float x4[kNE] = { cl[elem], cd[elem], relg[elem], mach[elem] };
  float al = alph[elem];

  // packed-f16 copy of u (static for the whole kernel) -> v_pk_mul_f16 operands
  h2 uh[4];
  #pragma unroll
  for (int q = 0; q < 4; ++q) {
    uh[q].x = (_Float16)uu[2 * q];
    uh[q].y = (_Float16)uu[2 * q + 1];
  }

  // ---------------- embed: tanh(Linear(2->D)) per expert ----------------
  float E[kNE][kD];
  #pragma unroll
  for (int i = 0; i < kNE; ++i)
    #pragma unroll
    for (int d = 0; d < kD; ++d) {
      float v = fmaf(We[(i * kD + d) * 2 + 0], x4[i],
                fmaf(We[(i * kD + d) * 2 + 1], al, be[i * kD + d]));
      E[i][d] = tanhf(v);
    }

  // ---------------- outer stages ----------------
  #pragma clang loop unroll(disable)
  for (int s = 0; s < kNS; ++s) {
    // A-fragments (16x32 f16): row m holds Wp[m%8][k]  -> rows 8-15 duplicate 0-7,
    // so both half-waves receive identical H in the D accumulator.
    const float* wpr = Wproj + s * kD * kD * kD + (col & 7) * (kD * kD);
    v16h af0, af1;
    #pragma unroll
    for (int j = 0; j < 16; ++j) {
      int kbase = hi * 16 + j;
      af0[j] = (_Float16)wpr[kbase];        // K in [0,32)
      af1[j] = (_Float16)wpr[32 + kbase];   // K in [32,64)
    }
    float cs[kNE][kD];
    #pragma unroll
    for (int i = 0; i < kNE; ++i)
      #pragma unroll
      for (int d = 0; d < kD; ++d) cs[i][d] = csbp[s][i][d];

    const float* W1 = Wr1 + s * kD * kD;   // wave-uniform -> scalar-load path
    const float* b1 = br1 + s * kD;
    const float* W2 = Wr2 + s * kNE * kD;
    const float* b2 = br2 + s * kNE;

    #pragma clang loop unroll(disable)
    for (int it = 0; it < kNI; ++it) {
      float H[kNE][kD];
      // ---- H[i] = (E_i (x) u) @ Wp^T + Cs_i  via 2 chained v_wmma_f32_16x16x32_f16 ----
      #pragma unroll
      for (int i = 0; i < kNE; ++i) {
        // This lane's 4 E-components feeding its B-matrix K-slices:
        //  kstep0: a = hi*2 + {0,1};  kstep1: a = 4 + hi*2 + {0,1}
        float e0f = hi ? E[i][2] : E[i][0];
        float e1f = hi ? E[i][3] : E[i][1];
        float e2f = hi ? E[i][6] : E[i][4];
        float e3f = hi ? E[i][7] : E[i][5];
        h2 e0; e0.x = (_Float16)e0f; e0.y = e0.x;
        h2 e1; e1.x = (_Float16)e1f; e1.y = e1.x;
        h2 e2; e2.x = (_Float16)e2f; e2.y = e2.x;
        h2 e3; e3.x = (_Float16)e3f; e3.y = e3.x;
        union V16 { v16h v; h2 h[8]; } b0, b1f;
        #pragma unroll
        for (int q = 0; q < 4; ++q) {       // packed f16 outer-product slices
          b0.h[q]      = e0 * uh[q];
          b0.h[4 + q]  = e1 * uh[q];
          b1f.h[q]     = e2 * uh[q];
          b1f.h[4 + q] = e3 * uh[q];
        }
        v8f c;
        #pragma unroll
        for (int d = 0; d < kD; ++d) c[d] = cs[i][d];
        c = __builtin_amdgcn_wmma_f32_16x16x32_f16(false, af0, false, b0.v,
                                                   (short)0, c, false, false);
        c = __builtin_amdgcn_wmma_f32_16x16x32_f16(false, af1, false, b1f.v,
                                                   (short)0, c, false, false);
        #pragma unroll
        for (int d = 0; d < kD; ++d) H[i][d] = c[d];
      }
      // ---- routing MLP + softmax over experts + mixture (per-lane f32 VALU) ----
      #pragma unroll
      for (int i = 0; i < kNE; ++i) {
        float G[kD];
        #pragma unroll
        for (int g = 0; g < kD; ++g) {
          float a = b1[g];
          #pragma unroll
          for (int d = 0; d < kD; ++d) a = fmaf(H[i][d], W1[g * kD + d], a);
          G[g] = fmaxf(a, 0.0f);
        }
        float lg[kNE], mx = -3.0e38f;
        #pragma unroll
        for (int e = 0; e < kNE; ++e) {
          float a = b2[e];
          #pragma unroll
          for (int g = 0; g < kD; ++g) a = fmaf(G[g], W2[e * kD + g], a);
          lg[e] = a; mx = fmaxf(mx, a);
        }
        float se = 0.0f;
        #pragma unroll
        for (int e = 0; e < kNE; ++e) { lg[e] = __expf(lg[e] - mx); se += lg[e]; }
        float inv = 1.0f / se;
        #pragma unroll
        for (int d = 0; d < kD; ++d) {
          float m = 0.0f;
          #pragma unroll
          for (int e = 0; e < kNE; ++e) m = fmaf(lg[e], H[e][d], m);
          E[i][d] = fmaf(m, inv, E[i][d]);
        }
      }
    }
    // ---- post-stage LayerNorm over d ----
    #pragma unroll
    for (int i = 0; i < kNE; ++i) {
      float m = 0.0f;
      #pragma unroll
      for (int d = 0; d < kD; ++d) m += E[i][d];
      m *= (1.0f / kD);
      float v = 0.0f;
      #pragma unroll
      for (int d = 0; d < kD; ++d) { float t = E[i][d] - m; v = fmaf(t, t, v); }
      float r = rsqrtf(v * (1.0f / kD) + kEPS);
      #pragma unroll
      for (int d = 0; d < kD; ++d)
        E[i][d] = fmaf((E[i][d] - m) * r, ln_g[d], ln_b[d]);
    }
  }

  // ---------------- final softmax mixing + LN + heads ----------------
  float lgo[kNE], mxo = -3.0e38f;
  #pragma unroll
  for (int n = 0; n < kNE; ++n) {
    float a = bout[n];
    #pragma unroll
    for (int i = 0; i < kNE; ++i)
      #pragma unroll
      for (int d = 0; d < kD; ++d)
        a = fmaf(E[i][d], Wout[n * (kNE * kD) + i * kD + d], a);
    lgo[n] = a; mxo = fmaxf(mxo, a);
  }
  float seo = 0.0f;
  #pragma unroll
  for (int n = 0; n < kNE; ++n) { lgo[n] = __expf(lgo[n] - mxo); seo += lgo[n]; }
  float invo = 1.0f / seo;

  float z[kD];
  #pragma unroll
  for (int d = 0; d < kD; ++d) {
    float a = 0.0f;
    #pragma unroll
    for (int n = 0; n < kNE; ++n) a = fmaf(lgo[n], E[n][d], a);
    z[d] = a * invo;
  }
  float m = 0.0f;
  #pragma unroll
  for (int d = 0; d < kD; ++d) m += z[d];
  m *= (1.0f / kD);
  float v = 0.0f;
  #pragma unroll
  for (int d = 0; d < kD; ++d) { float t = z[d] - m; v = fmaf(t, t, v); }
  float r = rsqrtf(v * (1.0f / kD) + kEPS);
  #pragma unroll
  for (int d = 0; d < kD; ++d) z[d] = fmaf((z[d] - m) * r, fg[d], fb[d]);

  if (hi == 0) {  // low half-wave writes the 19 outputs for its element
    float* o = out + (size_t)elem * 19;
    #pragma unroll
    for (int j = 0; j < 18; ++j) {
      float a = bcst[j];
      #pragma unroll
      for (int d = 0; d < kD; ++d) a = fmaf(z[d], Wcst[j * kD + d], a);
      o[j] = a;
    }
    float a = bcl[0];
    #pragma unroll
    for (int d = 0; d < kD; ++d) a = fmaf(z[d], Wcl[d], a);
    o[18] = a;
  }
}

extern "C" void kernel_launch(void* const* d_in, const int* in_sizes, int n_in,
                              void* d_out, int out_size, void* d_ws, size_t ws_size,
                              hipStream_t stream) {
  (void)n_in; (void)out_size; (void)d_ws; (void)ws_size;
  const int B = in_sizes[0];
  dim3 grid((B + kElemsPerBlock - 1) / kElemsPerBlock);
  dim3 block(256);
  whisp_fused<<<grid, block, 0, stream>>>(
      (const float*)d_in[0],  (const float*)d_in[1],  (const float*)d_in[2],
      (const float*)d_in[3],  (const float*)d_in[4],  (const float*)d_in[5],
      (const float*)d_in[6],  (const float*)d_in[7],  (const float*)d_in[8],
      (const float*)d_in[9],  (const float*)d_in[10], (const float*)d_in[11],
      (const float*)d_in[12], (const float*)d_in[13], (const float*)d_in[14],
      (const float*)d_in[15], (const float*)d_in[16], (const float*)d_in[17],
      (const float*)d_in[18], (const float*)d_in[19], (const float*)d_in[20],
      (const float*)d_in[21], (const float*)d_in[22], (const float*)d_in[23],
      (const float*)d_in[24], (float*)d_out, B);
}
